// Seq2SeqBiGRU_59124519796987
// MI455X (gfx1250) — matrix operand order repaired and checked by reading, not compiled, not run on hardware
//
#include <hip/hip_runtime.h>
#include <hip/hip_bf16.h>
#include <math.h>

typedef __bf16 bf16;
typedef bf16  v16bf __attribute__((ext_vector_type(16)));
typedef bf16  v8bf  __attribute__((ext_vector_type(8)));
typedef float v8f   __attribute__((ext_vector_type(8)));

#define B_    256
#define T_    512
#define F_    64
#define H_    256
#define DH_   512
#define OUT_  2
#define OSL_  20
#define NWG_  16
#define WAVES_PER_BLOCK 8
#define BLOCK_THREADS   256
#define LDS_PAD 8              // +8 bf16 per row -> conflict-free b128 reads

// ---------------------------------------------------------------------------
// WMMA helpers (CDNA5 wave32, V_WMMA_F32_16X16X32_BF16)
// ---------------------------------------------------------------------------

__device__ __forceinline__ v8f vzero8() {
    v8f z;
#pragma unroll
    for (int i = 0; i < 8; ++i) z[i] = 0.0f;
    return z;
}

// A matrix 16x32 bf16 (ISA 7.12.2): lane L(0..15) holds row M=L,
// K = {0..7, 16..23}; lanes 16..31 hold K = {8..15, 24..31}.
__device__ __forceinline__ v16bf load_a_tile(const bf16* __restrict__ base,
                                             int ld, int m0, int k0, int lane) {
    const int row = m0 + (lane & 15);
    const int hi  = (lane >> 4) & 1;
    const bf16* p = base + (size_t)row * ld + k0 + hi * 8;
    v8bf lo = *reinterpret_cast<const v8bf*>(p);
    v8bf hh = *reinterpret_cast<const v8bf*>(p + 16);
    v16bf a;
#pragma unroll
    for (int i = 0; i < 8; ++i) { a[i] = lo[i]; a[i + 8] = hh[i]; }
    return a;
}

// B matrix 32x16 bf16: lane L holds column N=L&15, K = {0..7, 16..23} (lanes 0-15)
// or {8..15, 24..31} (lanes 16-31).  B(k,n) = W[n,k], W row-major: two 8-bf16
// runs per lane.  Works for both global and LDS pointers (addrspace inferred).
__device__ __forceinline__ v16bf load_b_tile(const bf16* W, int ld,
                                             int n0, int k0, int lane) {
    const bf16* p = W + (size_t)(n0 + (lane & 15)) * ld + k0 +
                    ((lane >> 4) & 1) * 8;
    v8bf lo = *reinterpret_cast<const v8bf*>(p);
    v8bf hh = *reinterpret_cast<const v8bf*>(p + 16);
    v16bf b;
#pragma unroll
    for (int i = 0; i < 8; ++i) { b[i] = lo[i]; b[i + 8] = hh[i]; }
    return b;
}

__device__ __forceinline__ v8f wmma_bf16(v16bf a, v16bf b, v8f c) {
    return __builtin_amdgcn_wmma_f32_16x16x32_bf16(
        /*neg_a=*/false, a, /*neg_b=*/false, b,
        /*c_mod=*/(short)0, c, /*reuse_a=*/false, /*reuse_b=*/false);
}

__device__ __forceinline__ float fsigmoid(float x) {
    return 1.0f / (1.0f + __expf(-x));
}

// ---------------------------------------------------------------------------
// Device-scope generation barrier across the co-resident workgroup group.
// ---------------------------------------------------------------------------
__device__ __forceinline__ void grid_barrier(unsigned* bar, unsigned nblocks) {
    __threadfence();
    __syncthreads();
    if (threadIdx.x == 0) {
        unsigned ticket = __hip_atomic_fetch_add(bar, 1u, __ATOMIC_ACQ_REL,
                                                 __HIP_MEMORY_SCOPE_AGENT);
        unsigned target = (ticket / nblocks + 1u) * nblocks;
        while (__hip_atomic_load(bar, __ATOMIC_ACQUIRE,
                                 __HIP_MEMORY_SCOPE_AGENT) < target) {
            __builtin_amdgcn_s_sleep(2);
        }
    }
    __syncthreads();
}

// ---------------------------------------------------------------------------
// Stage this WG's weight slice (rows {g*Hg + j0base .. +JR} for g=0..2,
// all K columns) from global into padded LDS.  Vectorized 16B copies.
// ---------------------------------------------------------------------------
__device__ __forceinline__ void stage_weights(const bf16* __restrict__ gsrc,
                                              int K, int Hg, int j0base, int JR,
                                              bf16* ldst, int ldw) {
    const int kv = K >> 3;                 // 8-element vectors per row
    const int total = 3 * JR * kv;
    for (int idx = threadIdx.x; idx < total; idx += BLOCK_THREADS) {
        const int r  = idx / kv;
        const int c8 = (idx - r * kv) << 3;
        const int g  = r / JR;
        const int jr = r - g * JR;
        *reinterpret_cast<v8bf*>(ldst + (size_t)r * ldw + c8) =
            *reinterpret_cast<const v8bf*>(
                gsrc + (size_t)(g * Hg + j0base + jr) * K + c8);
    }
}

// ---------------------------------------------------------------------------
// One GRU timestep for this WG's column slice (weights in LDS).
//   gates = x_t @ Wih.T + h @ Whh.T   (4 f32 WMMA accumulators)
//   h' = (1-z)*n + z*h,  n = tanh(ni + bih_n + r*(nh + bhh_n))
// ---------------------------------------------------------------------------
__device__ __forceinline__ void gru_step_slice(
    const bf16* __restrict__ xa, int ldx, int Kx,
    const bf16* lWih, int ldwi,             // LDS [3*JR, Kx] pitch ldwi
    const bf16* lWhh, int ldwh,             // LDS [3*JR, Hg] pitch ldwh
    const float* __restrict__ bih, const float* __restrict__ bhh,
    int Hg, int j0base, int JR,
    const bf16* __restrict__ hin, bf16* __restrict__ hout, int ldh, int hcol,
    bf16* __restrict__ yrow, int ldy, int ycol,
    int wave, int lane)
{
    for (int jj = 0; jj < JR; jj += 16) {
        for (int mt = wave; mt < (B_ >> 4); mt += WAVES_PER_BLOCK) {
            const int m0 = mt << 4;
            v8f ar = vzero8(), az = vzero8(), ani = vzero8(), anh = vzero8();

            for (int k = 0; k < Kx; k += 32) {
                v16bf a = load_a_tile(xa, ldx, m0, k, lane);
                ar  = wmma_bf16(a, load_b_tile(lWih, ldwi, 0 * JR + jj, k, lane), ar);
                az  = wmma_bf16(a, load_b_tile(lWih, ldwi, 1 * JR + jj, k, lane), az);
                ani = wmma_bf16(a, load_b_tile(lWih, ldwi, 2 * JR + jj, k, lane), ani);
            }
            for (int k = 0; k < Hg; k += 32) {
                v16bf a = load_a_tile(hin, ldh, m0, hcol + k, lane);
                ar  = wmma_bf16(a, load_b_tile(lWhh, ldwh, 0 * JR + jj, k, lane), ar);
                az  = wmma_bf16(a, load_b_tile(lWhh, ldwh, 1 * JR + jj, k, lane), az);
                anh = wmma_bf16(a, load_b_tile(lWhh, ldwh, 2 * JR + jj, k, lane), anh);
            }

            // C/D layout: lane l<16 -> D[i, j+l]; lane>=16 -> D[8+i, j+l-16]
            const int colg = j0base + jj + (lane & 15);   // global hidden col
            const int rb   = m0 + ((lane & 16) ? 8 : 0);
            const float br  = bih[colg]          + bhh[colg];
            const float bz  = bih[Hg + colg]     + bhh[Hg + colg];
            const float bni = bih[2 * Hg + colg];
            const float bnh = bhh[2 * Hg + colg];
#pragma unroll
            for (int i = 0; i < 8; ++i) {
                float r  = fsigmoid(ar[i] + br);
                float z  = fsigmoid(az[i] + bz);
                float n  = tanhf(ani[i] + bni + r * (anh[i] + bnh));
                float hp = (float)hin[(size_t)(rb + i) * ldh + hcol + colg];
                float hv = (1.0f - z) * n + z * hp;
                bf16 hb  = (bf16)hv;
                hout[(size_t)(rb + i) * ldh + hcol + colg] = hb;
                if (yrow) yrow[(size_t)(rb + i) * ldy + ycol + colg] = hb;
            }
        }
    }
}

// ---------------------------------------------------------------------------
// Encoder layer: grid (NWG j-slices x 2 directions), T sequential steps.
// WG owns hidden columns [blockIdx.x*16, +16) for the whole scan; its
// weight slice lives in LDS.  h double-buffered [B, 2H] (fwd cols 0..H-1,
// bwd cols H..2H-1) so final h == cat(hf, hb) for the decoder.
// ---------------------------------------------------------------------------
__global__ void __launch_bounds__(BLOCK_THREADS)
enc_layer_kernel(const bf16* __restrict__ xin, int ldx, long stepmul, int Kx,
                 const bf16* __restrict__ Wih_all, const bf16* __restrict__ Whh_all,
                 const float* __restrict__ bih_all, const float* __restrict__ bhh_all,
                 bf16* __restrict__ hA, bf16* __restrict__ hB,
                 bf16* __restrict__ yout, unsigned* __restrict__ bar)
{
    extern __shared__ char smem_raw[];
    const int dir    = blockIdx.y;
    const int j0base = blockIdx.x * 16;
    const int wave   = threadIdx.x >> 5;
    const int lane   = threadIdx.x & 31;

    const bf16*  Wih = Wih_all + (size_t)dir * 3 * H_ * Kx;
    const bf16*  Whh = Whh_all + (size_t)dir * 3 * H_ * H_;
    const float* bih = bih_all + (size_t)dir * 3 * H_;
    const float* bhh = bhh_all + (size_t)dir * 3 * H_;
    const int hcol = dir * H_;

    const int ldwi = Kx + LDS_PAD;
    const int ldwh = H_ + LDS_PAD;
    bf16* lWih = (bf16*)smem_raw;
    bf16* lWhh = (bf16*)(smem_raw + (size_t)3 * 16 * ldwi * sizeof(bf16));
    stage_weights(Wih, Kx, H_, j0base, 16, lWih, ldwi);
    stage_weights(Whh, H_, H_, j0base, 16, lWhh, ldwh);
    __syncthreads();

    for (int s = 0; s < T_; ++s) {
        const int t = dir ? (T_ - 1 - s) : s;
        const bf16* hin  = (s & 1) ? hB : hA;
        bf16*       hout = (s & 1) ? hA : hB;
        const bf16* xt   = xin + (size_t)t * stepmul;
        bf16* yrow = yout ? (yout + (size_t)t * B_ * DH_) : (bf16*)nullptr;

        // warm WGP$ with next step's input rows for this wave
        if (s + 1 < T_ && lane == 0) {
            const int tn = dir ? (T_ - 2 - s) : (s + 1);
            __builtin_prefetch(xin + (size_t)tn * stepmul +
                               (size_t)(wave * 32) * ldx, 0, 1);
        }

        gru_step_slice(xt, ldx, Kx, lWih, ldwi, lWhh, ldwh, bih, bhh, H_,
                       j0base, 16, hin, hout, DH_, hcol, yrow, DH_, hcol,
                       wave, lane);
        grid_barrier(bar, NWG_ * 2);   // both directions step in lockstep
    }
}

// ---------------------------------------------------------------------------
// Decoder: 20 autoregressive steps, 2 GRU layers (DH=512) + FC.
// WG owns 32 hidden columns; all four weight slices resident in LDS (~300KB).
// ---------------------------------------------------------------------------
__global__ void __launch_bounds__(BLOCK_THREADS)
dec_kernel(const bf16* __restrict__ Wih0, const bf16* __restrict__ Whh0,
           const float* __restrict__ bih0, const float* __restrict__ bhh0,
           const bf16* __restrict__ Wih1, const bf16* __restrict__ Whh1,
           const float* __restrict__ bih1, const float* __restrict__ bhh1,
           const float* __restrict__ fcW, const float* __restrict__ fcb,
           bf16* __restrict__ h0a, bf16* __restrict__ h0b,
           bf16* __restrict__ h1a, bf16* __restrict__ h1b,
           bf16* __restrict__ in0, bf16* __restrict__ in1,
           float* __restrict__ out, unsigned* __restrict__ bar)
{
    extern __shared__ char smem_raw[];
    const int j0base = blockIdx.x * 32;
    const int wave   = threadIdx.x >> 5;
    const int lane   = threadIdx.x & 31;

    const int ldwi0 = 32 + LDS_PAD;        // padded K for dec layer-0 input
    const int ldw   = DH_ + LDS_PAD;
    char* p = smem_raw;
    bf16* lWih0 = (bf16*)p; p += (size_t)3 * 32 * ldwi0 * sizeof(bf16);
    bf16* lWhh0 = (bf16*)p; p += (size_t)3 * 32 * ldw   * sizeof(bf16);
    bf16* lWih1 = (bf16*)p; p += (size_t)3 * 32 * ldw   * sizeof(bf16);
    bf16* lWhh1 = (bf16*)p;
    stage_weights(Wih0, 32,  DH_, j0base, 32, lWih0, ldwi0);
    stage_weights(Whh0, DH_, DH_, j0base, 32, lWhh0, ldw);
    stage_weights(Wih1, DH_, DH_, j0base, 32, lWih1, ldw);
    stage_weights(Whh1, DH_, DH_, j0base, 32, lWhh1, ldw);
    __syncthreads();

    for (int s = 0; s < OSL_; ++s) {
        const bf16* xin  = (s & 1) ? in1 : in0;
        bf16*       xnxt = (s & 1) ? in0 : in1;
        const bf16* h0i = (s & 1) ? h0b : h0a;  bf16* h0o = (s & 1) ? h0a : h0b;
        const bf16* h1i = (s & 1) ? h1b : h1a;  bf16* h1o = (s & 1) ? h1a : h1b;

        // layer 0: input = prev prediction padded [B,32]
        gru_step_slice(xin, 32, 32, lWih0, ldwi0, lWhh0, ldw, bih0, bhh0, DH_,
                       j0base, 32, h0i, h0o, DH_, 0, (bf16*)nullptr, 0, 0,
                       wave, lane);
        grid_barrier(bar, NWG_);

        // layer 1: input = h0'
        gru_step_slice(h0o, DH_, DH_, lWih1, ldw, lWhh1, ldw, bih1, bhh1, DH_,
                       j0base, 32, h1i, h1o, DH_, 0, (bf16*)nullptr, 0, 0,
                       wave, lane);
        grid_barrier(bar, NWG_);

        // FC: pred[b,o] = h1' . fcW[o] + fcb[o]; feed back as next input
        const int gid = blockIdx.x * BLOCK_THREADS + threadIdx.x;
        if (gid < B_ * OUT_) {
            const int b = gid >> 1, o = gid & 1;
            float acc = fcb[o];
            for (int j = 0; j < DH_; ++j)
                acc += (float)h1o[(size_t)b * DH_ + j] * fcW[o * DH_ + j];
            out[(size_t)b * (OSL_ * OUT_) + s * OUT_ + o] = acc;
            xnxt[b * 32 + o] = (bf16)acc;
        }
        grid_barrier(bar, NWG_);
    }
}

// ---------------------------------------------------------------------------
// Prep kernels
// ---------------------------------------------------------------------------
__global__ void cvt_f32_bf16(const float* __restrict__ s, bf16* __restrict__ d,
                             long n) {
    long i = (long)blockIdx.x * blockDim.x + threadIdx.x;
    long stride = (long)gridDim.x * blockDim.x;
    for (; i < n; i += stride) d[i] = (bf16)s[i];
}

__global__ void pad_dec_wih0(const float* __restrict__ s, bf16* __restrict__ d) {
    int i = blockIdx.x * blockDim.x + threadIdx.x;   // over 3*DH_*32
    if (i < 3 * DH_ * 32) {
        int r = i >> 5, c = i & 31;
        d[i] = (c < OUT_) ? (bf16)s[r * OUT_ + c] : (bf16)0.0f;
    }
}

__global__ void init_start_pad(const float* __restrict__ st, bf16* __restrict__ d) {
    int i = blockIdx.x * blockDim.x + threadIdx.x;   // B_*OUT_
    if (i < B_ * OUT_) {
        int b = i >> 1, o = i & 1;
        d[b * 32 + o] = (bf16)st[o];
    }
}

// ---------------------------------------------------------------------------
// Host launcher
// ---------------------------------------------------------------------------
extern "C" void kernel_launch(void* const* d_in, const int* in_sizes, int n_in,
                              void* d_out, int out_size, void* d_ws, size_t ws_size,
                              hipStream_t stream)
{
    (void)in_sizes; (void)n_in; (void)out_size; (void)ws_size;

    const float* x     = (const float*)d_in[0];
    const float* start = (const float*)d_in[1];
    const float* eWih0 = (const float*)d_in[2];
    const float* eWhh0 = (const float*)d_in[3];
    const float* ebih0 = (const float*)d_in[4];
    const float* ebhh0 = (const float*)d_in[5];
    const float* eWih1 = (const float*)d_in[6];
    const float* eWhh1 = (const float*)d_in[7];
    const float* ebih1 = (const float*)d_in[8];
    const float* ebhh1 = (const float*)d_in[9];
    const float* dWih0 = (const float*)d_in[10];
    const float* dWhh0 = (const float*)d_in[11];
    const float* dbih0 = (const float*)d_in[12];
    const float* dbhh0 = (const float*)d_in[13];
    const float* dWih1 = (const float*)d_in[14];
    const float* dWhh1 = (const float*)d_in[15];
    const float* dbih1 = (const float*)d_in[16];
    const float* dbhh1 = (const float*)d_in[17];
    const float* fcW   = (const float*)d_in[18];
    const float* fcb   = (const float*)d_in[19];
    float* out = (float*)d_out;
    char*  ws  = (char*)d_ws;

    size_t off = 0;
    auto alloc = [&](size_t bytes) -> size_t {
        off = (off + 255) & ~(size_t)255;
        size_t o = off; off += bytes; return o;
    };

    // ---- zeroed region ----
    size_t bar_off  = alloc(3 * sizeof(unsigned));
    size_t eh0a_off = alloc((size_t)B_ * DH_ * 2);   // enc L0 h ping
    size_t eh0b_off = alloc((size_t)B_ * DH_ * 2);   // enc L0 h pong
    size_t eh1a_off = alloc((size_t)B_ * DH_ * 2);   // enc L1 h ping
    size_t eh1b_off = alloc((size_t)B_ * DH_ * 2);   // enc L1 h pong
    size_t dh0b_off = alloc((size_t)B_ * DH_ * 2);   // dec L0 h pong
    size_t dh1b_off = alloc((size_t)B_ * DH_ * 2);   // dec L1 h pong
    size_t in0_off  = alloc((size_t)B_ * 32 * 2);    // dec input pad ping
    size_t in1_off  = alloc((size_t)B_ * 32 * 2);    // dec input pad pong
    size_t zero_bytes = off;

    // ---- bf16 data / weights ----
    size_t xbf_off   = alloc((size_t)B_ * T_ * F_ * 2);
    size_t y0_off    = alloc((size_t)T_ * B_ * DH_ * 2);     // enc L0 output
    size_t ewih0_off = alloc((size_t)2 * 3 * H_ * F_  * 2);
    size_t ewhh0_off = alloc((size_t)2 * 3 * H_ * H_  * 2);
    size_t ewih1_off = alloc((size_t)2 * 3 * H_ * DH_ * 2);
    size_t ewhh1_off = alloc((size_t)2 * 3 * H_ * H_  * 2);
    size_t dwih0_off = alloc((size_t)3 * DH_ * 32  * 2);     // padded K=32
    size_t dwhh0_off = alloc((size_t)3 * DH_ * DH_ * 2);
    size_t dwih1_off = alloc((size_t)3 * DH_ * DH_ * 2);
    size_t dwhh1_off = alloc((size_t)3 * DH_ * DH_ * 2);

    hipMemsetAsync(ws, 0, zero_bytes, stream);

    auto cvt = [&](const float* s, size_t dst, long n) {
        int blocks = (int)((n + 255) / 256);
        if (blocks > 4096) blocks = 4096;
        cvt_f32_bf16<<<blocks, 256, 0, stream>>>(s, (bf16*)(ws + dst), n);
    };
    cvt(x,     xbf_off,   (long)B_ * T_ * F_);
    cvt(eWih0, ewih0_off, 2L * 3 * H_ * F_);
    cvt(eWhh0, ewhh0_off, 2L * 3 * H_ * H_);
    cvt(eWih1, ewih1_off, 2L * 3 * H_ * DH_);
    cvt(eWhh1, ewhh1_off, 2L * 3 * H_ * H_);
    cvt(dWhh0, dwhh0_off, 3L * DH_ * DH_);
    cvt(dWih1, dwih1_off, 3L * DH_ * DH_);
    cvt(dWhh1, dwhh1_off, 3L * DH_ * DH_);
    pad_dec_wih0<<<(3 * DH_ * 32 + 255) / 256, 256, 0, stream>>>(
        dWih0, (bf16*)(ws + dwih0_off));
    init_start_pad<<<2, 256, 0, stream>>>(start, (bf16*)(ws + in0_off));

    unsigned* bars = (unsigned*)(ws + bar_off);

    // dynamic LDS sizes (bf16, padded rows)
    const size_t shm_enc0 = (size_t)(3 * 16 * (F_ + LDS_PAD) +
                                     3 * 16 * (H_ + LDS_PAD)) * 2;
    const size_t shm_enc1 = (size_t)(3 * 16 * (DH_ + LDS_PAD) +
                                     3 * 16 * (H_ + LDS_PAD)) * 2;
    const size_t shm_dec  = (size_t)(3 * 32 * (32 + LDS_PAD) +
                                     3 * 32 * (DH_ + LDS_PAD) * 3) * 2;

    // encoder layer 0: x is [B,T,F] -> per-step A base offset t*F, ld = T*F
    enc_layer_kernel<<<dim3(NWG_, 2), BLOCK_THREADS, shm_enc0, stream>>>(
        (const bf16*)(ws + xbf_off), T_ * F_, (long)F_, F_,
        (const bf16*)(ws + ewih0_off), (const bf16*)(ws + ewhh0_off),
        ebih0, ebhh0,
        (bf16*)(ws + eh0a_off), (bf16*)(ws + eh0b_off),
        (bf16*)(ws + y0_off), bars + 0);

    // encoder layer 1: y0 is [T,B,2H] -> per-step offset t*B*DH, ld = DH
    enc_layer_kernel<<<dim3(NWG_, 2), BLOCK_THREADS, shm_enc1, stream>>>(
        (const bf16*)(ws + y0_off), DH_, (long)B_ * DH_, DH_,
        (const bf16*)(ws + ewih1_off), (const bf16*)(ws + ewhh1_off),
        ebih1, ebhh1,
        (bf16*)(ws + eh1a_off), (bf16*)(ws + eh1b_off),
        (bf16*)nullptr, bars + 1);

    // decoder: initial hiddens = encoder final h (land in the "a" buffers,
    // already laid out as cat(h_fwd, h_bwd) across columns)
    dec_kernel<<<NWG_, BLOCK_THREADS, shm_dec, stream>>>(
        (const bf16*)(ws + dwih0_off), (const bf16*)(ws + dwhh0_off), dbih0, dbhh0,
        (const bf16*)(ws + dwih1_off), (const bf16*)(ws + dwhh1_off), dbih1, dbhh1,
        fcW, fcb,
        (bf16*)(ws + eh0a_off), (bf16*)(ws + dh0b_off),
        (bf16*)(ws + eh1a_off), (bf16*)(ws + dh1b_off),
        (bf16*)(ws + in0_off), (bf16*)(ws + in1_off),
        out, bars + 2);
}
